// VGG_SMALL_1W1A_88596585382501
// MI455X (gfx1250) — 8-bit matrix operand order repaired and checked by reading, compiled, not run on hardware
//
#include <hip/hip_runtime.h>
#include <cstdint>
#include <cstddef>


typedef __attribute__((ext_vector_type(8))) int v8i;

#define EPSV 1e-5f

__device__ __forceinline__ int8_t sgn8(float v) {
    return (v > 0.f) ? (int8_t)1 : ((v < 0.f) ? (int8_t)-1 : (int8_t)0);
}

__global__ __launch_bounds__(256) void fill_zero16(uint4* __restrict__ p, size_t n16) {
    size_t i = (size_t)blockIdx.x * 256 + threadIdx.x;
    if (i < n16) {
        uint4 z; z.x = 0u; z.y = 0u; z.z = 0u; z.w = 0u;
        p[i] = z;
    }
}

__global__ __launch_bounds__(256)
void prep_weights(const float* __restrict__ w, int8_t* __restrict__ wq,
                  float* __restrict__ alpha, int Cin) {
    const int cout = blockIdx.x;
    const int K = Cin * 9;
    const float* wsrc = w + (size_t)cout * K;
    int8_t* wdst = wq + (size_t)cout * K;
    float sum = 0.f;
    for (int s = threadIdx.x; s < K; s += 256) {
        int ci = s / 9, r = s - ci * 9;
        int ky = r / 3, kx = r - ky * 3;
        float v = wsrc[s];
        sum += fabsf(v);
        wdst[(ky * 3 + kx) * Cin + ci] = sgn8(v);
    }
    __shared__ float red[256];
    red[threadIdx.x] = sum;
    __syncthreads();
    for (int sft = 128; sft > 0; sft >>= 1) {
        if (threadIdx.x < sft) red[threadIdx.x] += red[threadIdx.x + sft];
        __syncthreads();
    }
    if (threadIdx.x == 0) alpha[cout] = red[0] / (float)K;
}

__global__ __launch_bounds__(128)
void conv0_bn_sign(const float* __restrict__ x, const float* __restrict__ w0,
                   const float* __restrict__ g0, const float* __restrict__ b0,
                   const float* __restrict__ m0, const float* __restrict__ v0,
                   int8_t* __restrict__ act0) {
    const int co = threadIdx.x;
    const int pix = blockIdx.x;
    const int n = pix >> 10;
    const int y = (pix >> 5) & 31;
    const int xx = pix & 31;

    __shared__ float xin[27];
    if (threadIdx.x < 27) {
        int ci = threadIdx.x / 9, r = threadIdx.x - ci * 9;
        int ky = r / 3, kx = r - ky * 3;
        int iy = y + ky - 1, ix = xx + kx - 1;
        float v = 0.f;
        if (iy >= 0 && iy < 32 && ix >= 0 && ix < 32)
            v = x[(((size_t)n * 3 + ci) * 32 + iy) * 32 + ix];
        xin[threadIdx.x] = v;
    }
    __syncthreads();

    float acc = 0.f;
#pragma unroll
    for (int t = 0; t < 27; ++t)
        acc = fmaf(xin[t], w0[co * 27 + t], acc);

    float inv = g0[co] * rsqrtf(v0[co] + EPSV);
    float val = acc * inv + (b0[co] - m0[co] * inv);
    act0[(((size_t)n * 34 + y + 1) * 34 + (xx + 1)) * 128 + co] = sgn8(val);
}

template <int CIN, int COUT, int HH, int WW>
__global__ __launch_bounds__(128)
void binconv_wmma(const int8_t* __restrict__ actIn, const int8_t* __restrict__ wq,
                  const float* __restrict__ alpha, float* __restrict__ convOut) {
    constexpr int P = HH * WW;
    constexpr int Wp = WW + 2;
    constexpr int Hp = HH + 2;
    constexpr int CPK = (3 * CIN) / 64;
    constexpr int KTOT = 9 * CIN;

    const int lane = threadIdx.x & 31;
    const int wave = threadIdx.x >> 5;
    const int g = lane >> 4;
    const int mrow = lane & 15;

    const int pix0 = blockIdx.x * 16;
    const int n = pix0 / P;
    const int p0 = pix0 - n * P;

    const int p = p0 + mrow;
    const int y = p / WW;
    const int x = p - y * WW;
    const int8_t* arow =
        actIn + ((size_t)(n * Hp + y) * Wp + x) * CIN + g * 16;

    const int c0 = blockIdx.y * 128 + wave * 32 + mrow;
    const int8_t* wrow0 = wq + (size_t)c0 * KTOT + g * 16;
    const int8_t* wrow1 = wrow0 + (size_t)16 * KTOT;

    __builtin_prefetch(wrow0, 0, 1);
    __builtin_prefetch(wrow1, 0, 1);

    v8i acc0 = {0, 0, 0, 0, 0, 0, 0, 0};
    v8i acc1 = {0, 0, 0, 0, 0, 0, 0, 0};

    int kb = 0;
#pragma unroll
    for (int ky = 0; ky < 3; ++ky) {
        const int8_t* pa = arow + (size_t)ky * (Wp * CIN);
#pragma unroll 2
        for (int c = 0; c < CPK; ++c) {
            uint4 a0 = *(const uint4*)(pa);
            uint4 a1 = *(const uint4*)(pa + 32);
            v8i A;
            A[0] = (int)a0.x; A[1] = (int)a0.y;
            A[2] = (int)a0.z; A[3] = (int)a0.w;
            A[4] = (int)a1.x; A[5] = (int)a1.y;
            A[6] = (int)a1.z; A[7] = (int)a1.w;

            const int8_t* pb0 = wrow0 + kb;
            uint4 b00 = *(const uint4*)(pb0);
            uint4 b01 = *(const uint4*)(pb0 + 32);
            v8i B0;
            B0[0] = (int)b00.x; B0[1] = (int)b00.y;
            B0[2] = (int)b00.z; B0[3] = (int)b00.w;
            B0[4] = (int)b01.x; B0[5] = (int)b01.y;
            B0[6] = (int)b01.z; B0[7] = (int)b01.w;

            const int8_t* pb1 = wrow1 + kb;
            uint4 b10 = *(const uint4*)(pb1);
            uint4 b11 = *(const uint4*)(pb1 + 32);
            v8i B1;
            B1[0] = (int)b10.x; B1[1] = (int)b10.y;
            B1[2] = (int)b10.z; B1[3] = (int)b10.w;
            B1[4] = (int)b11.x; B1[5] = (int)b11.y;
            B1[6] = (int)b11.z; B1[7] = (int)b11.w;

            acc0 = __builtin_amdgcn_wmma_i32_16x16x64_iu8(
                true, A, true, B0, acc0, false, false);
            acc1 = __builtin_amdgcn_wmma_i32_16x16x64_iu8(
                true, A, true, B1, acc1, false, false);

            pa += 64;
            kb += 64;
        }
    }

    const float av0 = alpha[c0];
    const float av1 = alpha[c0 + 16];
    float* ob = convOut + ((size_t)n * P + p0) * COUT;
#pragma unroll
    for (int j = 0; j < 8; ++j) {
        const size_t rowoff = (size_t)(j + g * 8) * COUT;
        ob[rowoff + c0] = av0 * (float)acc0[j];
        ob[rowoff + c0 + 16] = av1 * (float)acc1[j];
    }
}

template <bool POOL>
__global__ __launch_bounds__(256)
void pool_bn_bin(const float* __restrict__ conv, int8_t* __restrict__ act,
                 const float* __restrict__ gg, const float* __restrict__ bb,
                 const float* __restrict__ mm, const float* __restrict__ vv,
                 int H, int W, int C, int total) {
    int idx = blockIdx.x * 256 + threadIdx.x;
    if (idx >= total) return;
    const int Ho = POOL ? (H >> 1) : H;
    const int Wo = POOL ? (W >> 1) : W;
    int c = idx % C;
    int t1 = idx / C;
    int xo = t1 % Wo;
    int t2 = t1 / Wo;
    int yo = t2 % Ho;
    int nn = t2 / Ho;

    float val;
    if (POOL) {
        const float* pb = conv + (((size_t)nn * H + 2 * yo) * W + 2 * xo) * C + c;
        float v00 = pb[0];
        float v01 = pb[(size_t)C];
        float v10 = pb[(size_t)W * C];
        float v11 = pb[(size_t)W * C + C];
        val = fmaxf(fmaxf(v00, v01), fmaxf(v10, v11));
    } else {
        val = conv[(((size_t)nn * H + yo) * W + xo) * C + c];
    }
    float inv = gg[c] * rsqrtf(vv[c] + EPSV);
    float bnv = val * inv + (bb[c] - mm[c] * inv);
    const int Wp = Wo + 2, Hp = Ho + 2;
    act[(((size_t)nn * Hp + yo + 1) * Wp + xo + 1) * C + c] = sgn8(bnv);
}

__global__ __launch_bounds__(256)
void pool_bn_htanh(const float* __restrict__ conv, float* __restrict__ out,
                   const float* __restrict__ gg, const float* __restrict__ bb,
                   const float* __restrict__ mm, const float* __restrict__ vv,
                   int total) {
    int idx = blockIdx.x * 256 + threadIdx.x;
    if (idx >= total) return;
    const int C = 512, H = 8, W = 8;
    int c = idx % C;
    int t1 = idx / C;
    int xo = t1 & 3;
    int t2 = t1 >> 2;
    int yo = t2 & 3;
    int nn = t2 >> 2;

    const float* pb = conv + (((size_t)nn * H + 2 * yo) * W + 2 * xo) * C + c;
    float v00 = pb[0];
    float v01 = pb[(size_t)C];
    float v10 = pb[(size_t)W * C];
    float v11 = pb[(size_t)W * C + C];
    float val = fmaxf(fmaxf(v00, v01), fmaxf(v10, v11));
    float inv = gg[c] * rsqrtf(vv[c] + EPSV);
    float bnv = val * inv + (bb[c] - mm[c] * inv);
    bnv = fminf(1.f, fmaxf(-1.f, bnv));
    out[(((size_t)nn * 4 + yo) * 4 + xo) * C + c] = bnv;
}

__global__ __launch_bounds__(64)
void fc_kernel(const float* __restrict__ a, const float* __restrict__ fcw,
               const float* __restrict__ fcb, float* __restrict__ out) {
    const int bI = blockIdx.x;
    const float* ab = a + (size_t)bI * 8192;
    __shared__ float red[10][64];
    const int t = threadIdx.x;
    for (int o = 0; o < 10; ++o) {
        float sum = 0.f;
        for (int k = t; k < 8192; k += 64) {
            int c = k >> 4, s = k & 15;
            sum += ab[s * 512 + c] * fcw[o * 8192 + k];
        }
        red[o][t] = sum;
    }
    __syncthreads();
    if (t < 10) {
        float sum = 0.f;
#pragma unroll 8
        for (int i = 0; i < 64; ++i) sum += red[t][i];
        out[bI * 10 + t] = sum + fcb[t];
    }
}

static inline void launch_fill(void* p, size_t bytes, hipStream_t s) {
    size_t n16 = bytes / 16;
    unsigned grid = (unsigned)((n16 + 255) / 256);
    hipLaunchKernelGGL(fill_zero16, dim3(grid), dim3(256), 0, s, (uint4*)p, n16);
}

extern "C" void kernel_launch(void* const* d_in, const int* in_sizes, int n_in,
                              void* d_out, int out_size, void* d_ws, size_t ws_size,
                              hipStream_t stream) {
    (void)in_sizes; (void)n_in; (void)out_size; (void)ws_size;

    const float* x = (const float*)d_in[0];
    const float *w[6], *gp[6], *bp[6], *mp[6], *vp[6];
    for (int i = 0; i < 6; ++i) {
        w[i]  = (const float*)d_in[1 + 5 * i];
        gp[i] = (const float*)d_in[2 + 5 * i];
        bp[i] = (const float*)d_in[3 + 5 * i];
        mp[i] = (const float*)d_in[4 + 5 * i];
        vp[i] = (const float*)d_in[5 + 5 * i];
    }
    const float* fcw = (const float*)d_in[31];
    const float* fcb = (const float*)d_in[32];
    float* out = (float*)d_out;

    char* ws = (char*)d_ws;
    int8_t* wq1 = (int8_t*)(ws + 0);
    int8_t* wq2 = (int8_t*)(ws + 147456);
    int8_t* wq3 = (int8_t*)(ws + 442368);
    int8_t* wq4 = (int8_t*)(ws + 1032192);
    int8_t* wq5 = (int8_t*)(ws + 2211840);
    float* al1 = (float*)(ws + 4571136);
    float* al2 = (float*)(ws + 4571648);
    float* al3 = (float*)(ws + 4572672);
    float* al4 = (float*)(ws + 4573696);
    float* al5 = (float*)(ws + 4575744);
    char* actA = ws + 4577792;
    char* actB = ws + 42457600;
    float* conv = (float*)(ws + 53074432);

    hipLaunchKernelGGL(prep_weights, dim3(128), dim3(256), 0, stream, w[1], wq1, al1, 128);
    hipLaunchKernelGGL(prep_weights, dim3(256), dim3(256), 0, stream, w[2], wq2, al2, 128);
    hipLaunchKernelGGL(prep_weights, dim3(256), dim3(256), 0, stream, w[3], wq3, al3, 256);
    hipLaunchKernelGGL(prep_weights, dim3(512), dim3(256), 0, stream, w[4], wq4, al4, 256);
    hipLaunchKernelGGL(prep_weights, dim3(512), dim3(256), 0, stream, w[5], wq5, al5, 512);

    launch_fill(actA, (size_t)256 * 34 * 34 * 128, stream);
    hipLaunchKernelGGL(conv0_bn_sign, dim3(256 * 32 * 32), dim3(128), 0, stream,
                       x, w[0], gp[0], bp[0], mp[0], vp[0], (int8_t*)actA);

    hipLaunchKernelGGL((binconv_wmma<128, 128, 32, 32>), dim3(16384, 1), dim3(128),
                       0, stream, (const int8_t*)actA, wq1, al1, conv);
    launch_fill(actB, (size_t)256 * 18 * 18 * 128, stream);
    hipLaunchKernelGGL((pool_bn_bin<true>), dim3((256 * 16 * 16 * 128 + 255) / 256),
                       dim3(256), 0, stream, conv, (int8_t*)actB,
                       gp[1], bp[1], mp[1], vp[1], 32, 32, 128, 256 * 16 * 16 * 128);

    hipLaunchKernelGGL((binconv_wmma<128, 256, 16, 16>), dim3(4096, 2), dim3(128),
                       0, stream, (const int8_t*)actB, wq2, al2, conv);
    launch_fill(actA, (size_t)256 * 18 * 18 * 256, stream);
    hipLaunchKernelGGL((pool_bn_bin<false>), dim3((256 * 16 * 16 * 256 + 255) / 256),
                       dim3(256), 0, stream, conv, (int8_t*)actA,
                       gp[2], bp[2], mp[2], vp[2], 16, 16, 256, 256 * 16 * 16 * 256);

    hipLaunchKernelGGL((binconv_wmma<256, 256, 16, 16>), dim3(4096, 2), dim3(128),
                       0, stream, (const int8_t*)actA, wq3, al3, conv);
    launch_fill(actB, (size_t)256 * 10 * 10 * 256, stream);
    hipLaunchKernelGGL((pool_bn_bin<true>), dim3((256 * 8 * 8 * 256 + 255) / 256),
                       dim3(256), 0, stream, conv, (int8_t*)actB,
                       gp[3], bp[3], mp[3], vp[3], 16, 16, 256, 256 * 8 * 8 * 256);

    hipLaunchKernelGGL((binconv_wmma<256, 512, 8, 8>), dim3(1024, 4), dim3(128),
                       0, stream, (const int8_t*)actB, wq4, al4, conv);
    launch_fill(actA, (size_t)256 * 10 * 10 * 512, stream);
    hipLaunchKernelGGL((pool_bn_bin<false>), dim3((256 * 8 * 8 * 512 + 255) / 256),
                       dim3(256), 0, stream, conv, (int8_t*)actA,
                       gp[4], bp[4], mp[4], vp[4], 8, 8, 512, 256 * 8 * 8 * 512);

    hipLaunchKernelGGL((binconv_wmma<512, 512, 8, 8>), dim3(1024, 4), dim3(128),
                       0, stream, (const int8_t*)actA, wq5, al5, conv);
    hipLaunchKernelGGL(pool_bn_htanh, dim3((256 * 4 * 4 * 512 + 255) / 256),
                       dim3(256), 0, stream, conv, (float*)actB,
                       gp[5], bp[5], mp[5], vp[5], 256 * 4 * 4 * 512);

    hipLaunchKernelGGL(fc_kernel, dim3(256), dim3(64), 0, stream,
                       (const float*)actB, fcw, fcb, out);
}
